// MSAColumnAttention_55688545960635
// MI455X (gfx1250) — matrix operand order repaired and checked by reading, not compiled, not run on hardware
//
#include <hip/hip_runtime.h>
#include <hip/hip_bf16.h>

// B=1, S=n_seq=256, L=seq_len=256, C_M=256, H=8, CH=32, HC=256
typedef __attribute__((ext_vector_type(16))) _Float16 v16h;
typedef __attribute__((ext_vector_type(8)))  _Float16 v8h;
typedef __attribute__((ext_vector_type(8)))  float    v8f;

union AV16 { v16h v; v8h h[2]; };

__device__ __forceinline__ v8f wmma16(v16h a, v16h b, v8f c) {
  // emits v_wmma_f32_16x16x32_f16
  return __builtin_amdgcn_wmma_f32_16x16x32_f16(false, a, false, b, (short)0, c, false, false);
}

// A-matrix 16x32 f16, row-major source. Per ISA 7.12.2:
// lanes 0-15 (M=lane): K=0..7 and K=16..23 ; lanes 16-31: K=8..15 and K=24..31
__device__ __forceinline__ v16h ldA(const _Float16* rowbase, int hi) {
  AV16 a;
  const _Float16* p = rowbase + hi * 8;
  a.h[0] = *(const v8h*)(p);
  a.h[1] = *(const v8h*)(p + 16);
  return a.v;
}

// B-matrix 32x16 f16 where column n is contiguous in memory (transposed buffer,
// row n holds K contiguously). lanes 0-15 (N=lane): K=0..7 and K=16..23 ; lanes 16-31: K=8..15 and K=24..31
__device__ __forceinline__ v16h ldB(const _Float16* base, int n, int ld, int hi) {
  AV16 b;
  const _Float16* p = base + (size_t)n * ld + hi * 8;
  b.h[0] = *(const v8h*)(p);
  b.h[1] = *(const v8h*)(p + 16);
  return b.v;
}

// ---------------- 1) LayerNorm over c_m, f32 -> f16 rows ----------------
__global__ void msa_ln_kernel(const float* __restrict__ m, const float* __restrict__ w,
                              const float* __restrict__ b, _Float16* __restrict__ mn) {
  const int row  = (blockIdx.x * blockDim.x + threadIdx.x) >> 5; // one wave per row
  const int lane = threadIdx.x & 31;
  const float* r = m + (size_t)row * 256;
  float vals[8];
  float s = 0.f;
#pragma unroll
  for (int i = 0; i < 8; ++i) { vals[i] = r[lane + 32 * i]; s += vals[i]; }
#pragma unroll
  for (int off = 16; off; off >>= 1) s += __shfl_xor(s, off, 32);
  const float mu = s * (1.f / 256.f);
  float vs = 0.f;
#pragma unroll
  for (int i = 0; i < 8; ++i) { float d = vals[i] - mu; vs += d * d; }
#pragma unroll
  for (int off = 16; off; off >>= 1) vs += __shfl_xor(vs, off, 32);
  const float inv = rsqrtf(vs * (1.f / 256.f) + 1e-5f);
  _Float16* o = mn + (size_t)row * 256;
#pragma unroll
  for (int i = 0; i < 8; ++i) {
    const int c = lane + 32 * i;
    o[c] = (_Float16)((vals[i] - mu) * inv * w[c] + b[c]);
  }
}

// ---------------- 2) weight prep: f32 [k][n] -> f16 transposed [n][k] ----------------
__global__ void msa_wprep_kernel(const float* __restrict__ wq, const float* __restrict__ wk,
                                 const float* __restrict__ wv, const float* __restrict__ wg,
                                 const float* __restrict__ wo, _Float16* __restrict__ wt) {
  const int mi = blockIdx.x >> 8;   // 0..4
  const int n  = blockIdx.x & 255;
  const int k  = threadIdx.x;
  const float* W = (mi == 0) ? wq : (mi == 1) ? wk : (mi == 2) ? wv : (mi == 3) ? wg : wo;
  wt[((size_t)mi << 16) + (size_t)n * 256 + k] = (_Float16)W[(size_t)k * 256 + n];
}

// ---------------- 3a) Q/K projection: rows (l,s) gathered from mn[s,l,:], out [l][h][s][c] -------
__global__ void msa_projqk_kernel(const _Float16* __restrict__ mn, const _Float16* __restrict__ wt,
                                  _Float16* __restrict__ q, _Float16* __restrict__ k) {
  __shared__ _Float16 T[4][16][256]; // 32 KB staging for coalesced stores
  const int wave = threadIdx.x >> 5;        // 0..3
  const int lane = threadIdx.x & 31;
  const int l15  = lane & 15;
  const int hi   = lane >> 4;
  const int which = blockIdx.y;             // 0=q 1=k
  const int mt = blockIdx.x * 4 + wave;     // 0..4095 m-tiles
  const int r0 = mt * 16;
  const int l  = r0 >> 8;                   // column index (constant within tile)
  const int s0 = r0 & 255;

  const _Float16* wbase = wt + ((size_t)which << 16);
  const _Float16* arow  = mn + ((size_t)(s0 + l15) * 256 + l) * 256; // row (l,s) -> mn[s*256+l]
  _Float16* outp = (which == 0) ? q : k;

  v16h a[8];
#pragma unroll
  for (int kk = 0; kk < 8; ++kk) a[kk] = ldA(arow + kk * 32, hi);

#pragma unroll 1
  for (int nt = 0; nt < 16; ++nt) {
    v8f acc = {};
#pragma unroll
    for (int kk = 0; kk < 8; ++kk) {
      v16h b = ldB(wbase + kk * 32, nt * 16 + l15, 256, hi);
      acc = wmma16(a[kk], b, acc);
    }
    const int col = nt * 16 + l15;
#pragma unroll
    for (int j = 0; j < 8; ++j) T[wave][j + hi * 8][col] = (_Float16)acc[j];
  }
  __syncthreads();
  // copy out: 16 s-rows x 8 heads x 64B segments -> b128 chunks (512 chunks, 16/lane)
#pragma unroll
  for (int t = 0; t < 16; ++t) {
    const int g    = lane + 32 * t;
    const int seg  = g >> 2;        // 0..127 : (h, s_local)
    const int ch   = g & 3;         // 16B chunk within 64B row segment
    const int h    = seg >> 4;
    const int sl   = seg & 15;
    v8h d = *(const v8h*)&T[wave][sl][h * 32 + ch * 8];
    *(v8h*)(outp + ((size_t)(l * 8 + h) * 256 + (s0 + sl)) * 32 + ch * 8) = d;
  }
}

// ---------------- 3b) V projection, stored transposed: out [l][h][c][s] ----------------
__global__ void msa_projv_kernel(const _Float16* __restrict__ mn, const _Float16* __restrict__ wt,
                                 _Float16* __restrict__ vt) {
  __shared__ _Float16 Tt[4][256][16]; // 32 KB, [col][s_local]
  const int wave = threadIdx.x >> 5;
  const int lane = threadIdx.x & 31;
  const int l15  = lane & 15;
  const int hi   = lane >> 4;
  const int mt = blockIdx.x * 4 + wave;
  const int r0 = mt * 16;
  const int l  = r0 >> 8;
  const int s0 = r0 & 255;

  const _Float16* wbase = wt + ((size_t)2 << 16);
  const _Float16* arow  = mn + ((size_t)(s0 + l15) * 256 + l) * 256;

  v16h a[8];
#pragma unroll
  for (int kk = 0; kk < 8; ++kk) a[kk] = ldA(arow + kk * 32, hi);

#pragma unroll 1
  for (int nt = 0; nt < 16; ++nt) {
    v8f acc = {};
#pragma unroll
    for (int kk = 0; kk < 8; ++kk) {
      v16h b = ldB(wbase + kk * 32, nt * 16 + l15, 256, hi);
      acc = wmma16(a[kk], b, acc);
    }
    const int col = nt * 16 + l15;
#pragma unroll
    for (int j = 0; j < 8; ++j) Tt[wave][col][j + hi * 8] = (_Float16)acc[j];
  }
  __syncthreads();
  // copy out: 256 (h,c) columns x 32B s-segments -> b128 chunks (512 chunks, 16/lane)
#pragma unroll
  for (int t = 0; t < 16; ++t) {
    const int g    = lane + 32 * t;
    const int col  = g >> 1;        // 0..255
    const int half = g & 1;         // which 16B of the 32B segment
    const int h = col >> 5, c = col & 31;
    v8h d = *(const v8h*)&Tt[wave][col][half * 8];
    *(v8h*)(vt + ((size_t)(l * 8 + h) * 32 + c) * 256 + s0 + half * 8) = d;
  }
}

// ---------------- 4) attention per (l,h,64-row block): logits->softmax->P@V->scrambled store --------
__global__ void msa_attn_kernel(const _Float16* __restrict__ q, const _Float16* __restrict__ k,
                                const _Float16* __restrict__ vt, _Float16* __restrict__ osc) {
  __shared__ _Float16 P[4][16][256];  // 32 KB probability rows (per wave)
  __shared__ _Float16 Ob[4][16][32];  // 4 KB output staging
  const int wave = threadIdx.x >> 5;  // 0..3
  const int lane = threadIdx.x & 31;
  const int l15  = lane & 15, hi = lane >> 4;
  const int h  = blockIdx.x & 7;
  const int mb = (blockIdx.x >> 3) & 3;
  const int l  = blockIdx.x >> 5;
  const int sbase = mb * 64 + wave * 16;

  const size_t lh = (size_t)(l * 8 + h);
  const _Float16* qb = q  + lh * 256 * 32;
  const _Float16* kb = k  + lh * 256 * 32;
  const _Float16* vb = vt + lh * 32 * 256;

  const v16h aq = ldA(qb + (size_t)(sbase + l15) * 32, hi);

  v8f acc[16];
#pragma unroll 1
  for (int nt = 0; nt < 16; ++nt) {
    v8f z = {};
    v16h b = ldB(kb, nt * 16 + l15, 32, hi); // K rows contiguous in c
    acc[nt] = wmma16(aq, b, z);
  }
  const float scale = 0.17677669529663687f; // 1/sqrt(32)
#pragma unroll
  for (int j = 0; j < 8; ++j) {
    float mx = -1e30f;
#pragma unroll
    for (int nt = 0; nt < 16; ++nt) { float x = acc[nt][j] * scale; acc[nt][j] = x; mx = fmaxf(mx, x); }
#pragma unroll
    for (int off = 1; off < 16; off <<= 1) mx = fmaxf(mx, __shfl_xor(mx, off, 32));
    float sum = 0.f;
#pragma unroll
    for (int nt = 0; nt < 16; ++nt) { float e = __expf(acc[nt][j] - mx); acc[nt][j] = e; sum += e; }
#pragma unroll
    for (int off = 1; off < 16; off <<= 1) sum += __shfl_xor(sum, off, 32);
    const float inv = 1.f / sum;
#pragma unroll
    for (int nt = 0; nt < 16; ++nt)
      P[wave][j + hi * 8][nt * 16 + l15] = (_Float16)(acc[nt][j] * inv);
  }
  __syncthreads();

#pragma unroll 1
  for (int ct = 0; ct < 2; ++ct) {
    v8f oa = {};
#pragma unroll
    for (int kk = 0; kk < 8; ++kk) {
      v16h a = ldA(&P[wave][l15][kk * 32], hi);
      v16h b = ldB(vb + kk * 32, ct * 16 + l15, 256, hi); // Vt rows contiguous in s
      oa = wmma16(a, b, oa);
    }
    const int c = ct * 16 + l15;
#pragma unroll
    for (int j = 0; j < 8; ++j) Ob[wave][j + hi * 8][c] = (_Float16)oa[j];
  }
  __syncthreads();
  // each source row (s fixed) is a contiguous 64B segment of the scrambled tensor:
  // osc[s' = h*32|(l>>3)][l' = ((l&7)<<5)|(s>>3)][hc' = ((s&7)<<5) + c], c=0..31
  const int sp = h * 32 + (l >> 3);
#pragma unroll
  for (int t = 0; t < 2; ++t) {
    const int g   = lane + 32 * t;  // 0..63
    const int row = g >> 2;         // s_local 0..15
    const int ch  = g & 3;          // 16B chunk within 64B
    const int s   = sbase + row;
    const int lp  = ((l & 7) << 5) | (s >> 3);
    v8h d = *(const v8h*)&Ob[wave][row][ch * 8];
    *(v8h*)(osc + (((size_t)sp * 256 + lp) << 8) + ((s & 7) << 5) + ch * 8) = d;
  }
}

// ---------------- 5) g = sigmoid(mn@wg+bg); out = (g*o)@wo + bo ----------------
__global__ void msa_gate_out_kernel(const _Float16* __restrict__ mn, const _Float16* __restrict__ wt,
                                    const _Float16* __restrict__ osc, const float* __restrict__ bg,
                                    const float* __restrict__ bo, float* __restrict__ out) {
  __shared__ _Float16 OG[4][16][256]; // 32 KB gated-o staging
  const int wave = threadIdx.x >> 5;  // 0..3
  const int lane = threadIdx.x & 31;
  const int l15  = lane & 15, hi = lane >> 4;
  const int mt = blockIdx.x * 4 + wave;
  const int r0 = mt * 16;             // rows r = s*256+l (natural order)
  const _Float16* wgT = wt + ((size_t)3 << 16);
  const _Float16* woT = wt + ((size_t)4 << 16);
  const _Float16* arow = mn + (size_t)(r0 + l15) * 256;

  v16h a[8];
#pragma unroll
  for (int kk = 0; kk < 8; ++kk) a[kk] = ldA(arow + kk * 32, hi);

#pragma unroll 1
  for (int nt = 0; nt < 16; ++nt) {
    v8f acc = {};
#pragma unroll
    for (int kk = 0; kk < 8; ++kk)
      acc = wmma16(a[kk], ldB(wgT + kk * 32, nt * 16 + l15, 256, hi), acc);
    const int col = nt * 16 + l15;
    const float bgc = bg[col];
#pragma unroll
    for (int j = 0; j < 8; ++j) {
      const int r = r0 + j + hi * 8;
      const float g = 1.f / (1.f + __expf(-(acc[j] + bgc)));
      const float o = (float)osc[(size_t)r * 256 + col];
      OG[wave][j + hi * 8][col] = (_Float16)(g * o);
    }
  }
  __syncthreads();

  v16h a2[8];
#pragma unroll
  for (int kk = 0; kk < 8; ++kk) a2[kk] = ldA(&OG[wave][l15][kk * 32], hi);

#pragma unroll 1
  for (int nt = 0; nt < 16; ++nt) {
    v8f acc = {};
#pragma unroll
    for (int kk = 0; kk < 8; ++kk)
      acc = wmma16(a2[kk], ldB(woT + kk * 32, nt * 16 + l15, 256, hi), acc);
    const int col = nt * 16 + l15;
    const float boc = bo[col];
#pragma unroll
    for (int j = 0; j < 8; ++j)
      out[(size_t)(r0 + j + hi * 8) * 256 + col] = acc[j] + boc; // 16 lanes x 4B coalesce
  }
}

extern "C" void kernel_launch(void* const* d_in, const int* in_sizes, int n_in,
                              void* d_out, int out_size, void* d_ws, size_t ws_size,
                              hipStream_t stream) {
  const float* m    = (const float*)d_in[0];
  const float* ln_w = (const float*)d_in[1];
  const float* ln_b = (const float*)d_in[2];
  const float* wq   = (const float*)d_in[3];
  const float* wk   = (const float*)d_in[4];
  const float* wv   = (const float*)d_in[5];
  const float* wg   = (const float*)d_in[6];
  const float* bg   = (const float*)d_in[7];
  const float* wo   = (const float*)d_in[8];
  const float* bo   = (const float*)d_in[9];
  float* out = (float*)d_out;

  char* ws = (char*)d_ws;
  const size_t MB32 = (size_t)32 << 20; // each f16 tensor is exactly 32 MiB
  _Float16* mn  = (_Float16*)(ws);               // [65536][256] layer-normed, f16
  _Float16* qb  = (_Float16*)(ws + 1 * MB32);    // [l][h][s][c]
  _Float16* kb  = (_Float16*)(ws + 2 * MB32);    // [l][h][s][c]
  _Float16* vtb = (_Float16*)(ws + 3 * MB32);    // [l][h][c][s]
  _Float16* osc = (_Float16*)(ws + 4 * MB32);    // [s][l][hc] scrambled o
  _Float16* wt  = (_Float16*)(ws + 5 * MB32);    // 5x [256][256] f16 transposed weights

  msa_ln_kernel<<<8192, 256, 0, stream>>>(m, ln_w, ln_b, mn);
  msa_wprep_kernel<<<1280, 256, 0, stream>>>(wq, wk, wv, wg, wo, wt);
  msa_projqk_kernel<<<dim3(1024, 2), 128, 0, stream>>>(mn, wt, qb, kb);
  msa_projv_kernel<<<1024, 128, 0, stream>>>(mn, wt, vtb);
  msa_attn_kernel<<<8192, 128, 0, stream>>>(qb, kb, vtb, osc);
  msa_gate_out_kernel<<<1024, 128, 0, stream>>>(mn, wt, osc, bg, bo, out);
}
